// shotLevelExtractionOverlap_79431125172929
// MI455X (gfx1250) — matrix operand order repaired and checked by reading, not compiled, not run on hardware
//
#include <hip/hip_runtime.h>

typedef __attribute__((ext_vector_type(16))) _Float16 v16h;
typedef __attribute__((ext_vector_type(8)))  float    v8f;
typedef __attribute__((ext_vector_type(4)))  float    v4f;

#define SEQ_LEN 8192
#define IN_CH   1024
#define SHOT_CH 8096
#define OVL     10240   // length of overlap-index sequence
#define NBINS   48      // 16 + 24 + 8
#define CATW    8094    // 2024 + 2023 + 2023 + 2024

// ---- overlap index map: r in [0,10240) -> original seq row -------------
__device__ __forceinline__ int ov_map(int r) {
    int i = r / 640, j = r - i * 640;   // each chunk = 128 overlap + 512 main
    if (j < 128) return (i == 0) ? (8064 + j) : (i * 512 - 128 + j);
    return i * 512 + (j - 128);
}

// ---- Kernel A: per-branch adaptive-average over ov rows -> meanrows[48,1024]
// seq (33.5 MB) is L2-resident (192 MB L2), so the 3x branch re-read is cheap.
__global__ void k_binmean(const float* __restrict__ seq, float* __restrict__ meanrows) {
    int g = blockIdx.x;                 // global bin 0..47
    int S, s;
    if (g < 16)      { S = 16; s = g;      }
    else if (g < 40) { S = 24; s = g - 16; }
    else             { S = 8;  s = g - 40; }
    int start = (s * OVL) / S;
    int end   = ((s + 1) * OVL + S - 1) / S;
    float inv = 1.0f / (float)(end - start);
    int c = blockIdx.y * 256 + threadIdx.x;     // channel 0..1023
    float acc = 0.0f;
    for (int r = start; r < end; ++r)
        acc += seq[(size_t)ov_map(r) * IN_CH + c];
    meanrows[(size_t)g * IN_CH + c] = acc * inv;
}

// ---- pack 16 f32 -> v16h fragment ---------------------------------------
__device__ __forceinline__ v16h pack16(v4f x0, v4f x1, v4f x2, v4f x3) {
    v16h h;
    h[0]=(_Float16)x0[0]; h[1]=(_Float16)x0[1]; h[2]=(_Float16)x0[2]; h[3]=(_Float16)x0[3];
    h[4]=(_Float16)x1[0]; h[5]=(_Float16)x1[1]; h[6]=(_Float16)x1[2]; h[7]=(_Float16)x1[3];
    h[8]=(_Float16)x2[0]; h[9]=(_Float16)x2[1]; h[10]=(_Float16)x2[2]; h[11]=(_Float16)x2[3];
    h[12]=(_Float16)x3[0]; h[13]=(_Float16)x3[1]; h[14]=(_Float16)x3[2]; h[15]=(_Float16)x3[3];
    return h;
}

// Load an A fragment (16x32 f16 layout) from a row-major f32 row pointer.
// lane m = row, halves j<8 -> K=kb+hi*8+j ; j>=8 -> K=kb+16+hi*8+(j-8)
__device__ __forceinline__ v16h load_a_frag(const float* Arow, int kb, int hi) {
    v4f a0 = *(const v4f*)(Arow + kb + hi * 8);
    v4f a1 = *(const v4f*)(Arow + kb + hi * 8 + 4);
    v4f a2 = *(const v4f*)(Arow + kb + 16 + hi * 8);
    v4f a3 = *(const v4f*)(Arow + kb + 16 + hi * 8 + 4);
    return pack16(a0, a1, a2, a3);
}

// Load a B fragment (32x16 f16 layout) from a row-major f32 row (col nCol's row).
// lane n = col, halves j<8 -> K=kb+hi*8+j ; j>=8 -> K=kb+16+hi*8+(j-8). Non-temporal: weights are
// single-use streams; keep L2 for seq/shot/pooled.
__device__ __forceinline__ v16h load_b_frag_nt(const float* Brow, int kb, int hi) {
    v4f b0 = __builtin_nontemporal_load((const v4f*)(Brow + kb + hi * 8));
    v4f b1 = __builtin_nontemporal_load((const v4f*)(Brow + kb + hi * 8 + 4));
    v4f b2 = __builtin_nontemporal_load((const v4f*)(Brow + kb + 16 + hi * 8));
    v4f b3 = __builtin_nontemporal_load((const v4f*)(Brow + kb + 16 + hi * 8 + 4));
    return pack16(b0, b1, b2, b3);
}

// ---- Kernel B: shot = meanrows @ w_up.T + b_up  via WMMA f16 ------------
// grid (506), block 32. One wave per N-tile computes ALL THREE M-tiles so each
// B fragment (w_up) is loaded exactly once -> w_up streamed 1x (33 MB, not 99).
__global__ void k_shot_gemm(const float* __restrict__ meanrows,
                            const float* __restrict__ w_up,
                            const float* __restrict__ b_up,
                            float* __restrict__ shot) {
    int lane = threadIdx.x;
    int m16  = lane & 15;
    int hi   = lane >> 4;
    int nt = blockIdx.x;
    const float* A0 = meanrows + (size_t)(m16)      * IN_CH;
    const float* A1 = meanrows + (size_t)(16 + m16) * IN_CH;
    const float* A2 = meanrows + (size_t)(32 + m16) * IN_CH;
    int nCol = nt * 16 + m16;
    const float* Brow = w_up + (size_t)nCol * IN_CH;
    v8f acc0 = {}, acc1 = {}, acc2 = {};
    for (int kb = 0; kb < IN_CH; kb += 32) {
        __builtin_prefetch(Brow + kb + 128, 0, 0);
        v16h b = load_b_frag_nt(Brow, kb, hi);
        v16h a0 = load_a_frag(A0, kb, hi);
        v16h a1 = load_a_frag(A1, kb, hi);
        v16h a2 = load_a_frag(A2, kb, hi);
        acc0 = __builtin_amdgcn_wmma_f32_16x16x32_f16(false, a0, false, b,
                                                      (short)0, acc0, false, false);
        acc1 = __builtin_amdgcn_wmma_f32_16x16x32_f16(false, a1, false, b,
                                                      (short)0, acc1, false, false);
        acc2 = __builtin_amdgcn_wmma_f32_16x16x32_f16(false, a2, false, b,
                                                      (short)0, acc2, false, false);
    }
    float bias = b_up[nCol];
    #pragma unroll
    for (int v = 0; v < 8; ++v) {
        int mLocal = v + hi * 8;        // D layout: VGPR v holds M = v + hi*8
        int col = nt * 16 + m16;
        shot[(size_t)(mLocal)      * SHOT_CH + col] = acc0[v] + bias;
        shot[(size_t)(16 + mLocal) * SHOT_CH + col] = acc1[v] + bias;
        shot[(size_t)(32 + mLocal) * SHOT_CH + col] = acc2[v] + bias;
    }
}

// ---- Kernel C: conv1d x4 (stride4, dil 1/2/3/1) + cat + upsample pool ---
// grid (64, 3), block 128. pooled[48, 8096]. shot (1.55 MB) is L2-resident.
__global__ void k_conv_pool(const float* __restrict__ shot,
    const float* __restrict__ w11, const float* __restrict__ w12,
    const float* __restrict__ w13, const float* __restrict__ w14,
    const float* __restrict__ c11, const float* __restrict__ c12,
    const float* __restrict__ c13, const float* __restrict__ c14,
    const float* __restrict__ w21, const float* __restrict__ w22,
    const float* __restrict__ w23, const float* __restrict__ w24,
    const float* __restrict__ c21, const float* __restrict__ c22,
    const float* __restrict__ c23, const float* __restrict__ c24,
    const float* __restrict__ w31, const float* __restrict__ w32,
    const float* __restrict__ w33, const float* __restrict__ w34,
    const float* __restrict__ c31, const float* __restrict__ c32,
    const float* __restrict__ c33, const float* __restrict__ c34,
    float* __restrict__ pooled) {
    int br = blockIdx.y;
    int C  = (br == 0) ? 16 : ((br == 1) ? 24 : 8);
    int rb = (br == 0) ? 0  : ((br == 1) ? 16 : 40);
    const float *W1, *W2, *W3, *W4, *B1, *B2, *B3, *B4;
    if (br == 0) { W1=w11; W2=w12; W3=w13; W4=w14; B1=c11; B2=c12; B3=c13; B4=c14; }
    else if (br == 1) { W1=w21; W2=w22; W3=w23; W4=w24; B1=c21; B2=c22; B3=c23; B4=c24; }
    else { W1=w31; W2=w32; W3=w33; W4=w34; B1=c31; B2=c32; B3=c33; B4=c34; }

    int j = blockIdx.x * 128 + threadIdx.x;
    if (j >= SHOT_CH) return;
    int start = (j * CATW) / SHOT_CH;
    int end   = ((j + 1) * CATW + SHOT_CH - 1) / SHOT_CH;
    float inv = 1.0f / (float)(end - start);

    for (int oc = 0; oc < C; ++oc) {
        float accp = 0.0f;
        for (int p = start; p < end; ++p) {
            float s;
            if (p < 2024) {                         // k=3, dil=1
                int iw = p * 4; s = B1[oc];
                for (int ic = 0; ic < C; ++ic) {
                    const float* xr = shot + (size_t)(rb + ic) * SHOT_CH + iw;
                    const float* wr = W1 + (oc * C + ic) * 3;
                    s += wr[0]*xr[0] + wr[1]*xr[1] + wr[2]*xr[2];
                }
            } else if (p < 4047) {                  // k=3, dil=2
                int iw = (p - 2024) * 4; s = B2[oc];
                for (int ic = 0; ic < C; ++ic) {
                    const float* xr = shot + (size_t)(rb + ic) * SHOT_CH + iw;
                    const float* wr = W2 + (oc * C + ic) * 3;
                    s += wr[0]*xr[0] + wr[1]*xr[2] + wr[2]*xr[4];
                }
            } else if (p < 6070) {                  // k=3, dil=3
                int iw = (p - 4047) * 4; s = B3[oc];
                for (int ic = 0; ic < C; ++ic) {
                    const float* xr = shot + (size_t)(rb + ic) * SHOT_CH + iw;
                    const float* wr = W3 + (oc * C + ic) * 3;
                    s += wr[0]*xr[0] + wr[1]*xr[3] + wr[2]*xr[6];
                }
            } else {                                // k=1
                int iw = (p - 6070) * 4; s = B4[oc];
                for (int ic = 0; ic < C; ++ic)
                    s += W4[oc * C + ic] * shot[(size_t)(rb + ic) * SHOT_CH + iw];
            }
            accp += s;
        }
        pooled[(size_t)(rb + oc) * SHOT_CH + j] = accp * inv;
    }
}

// ---- Kernel D: h = relu(pooled @ fd_w.T + fd_b)  via WMMA f16 -----------
// grid (64, 3), block 32. blockIdx.y = branch. Branch 1 (24 rows) computes two
// M-tiles (rows 16-31 and 32-47, 8 valid) sharing each fd2_w B fragment, so
// every fd*_w matrix is streamed exactly once. Branch selection is uniform per
// block, so EXEC stays all-ones at every WMMA.
__global__ void k_fd_gemm(const float* __restrict__ pooled,
                          const float* __restrict__ fd1_w, const float* __restrict__ fd2_w,
                          const float* __restrict__ fd3_w,
                          const float* __restrict__ fd1_b, const float* __restrict__ fd2_b,
                          const float* __restrict__ fd3_b,
                          float* __restrict__ h) {
    int lane = threadIdx.x, m16 = lane & 15, hi = lane >> 4;
    int nt = blockIdx.x, br = blockIdx.y;
    int row0   = (br == 0) ? 0 : ((br == 1) ? 16 : 40);
    int valid0 = (br == 2) ? 8 : 16;     // valid rows in first M-tile
    bool twoTiles = (br == 1);           // second tile: rows 32..39 (8 valid)
    const float* fw = (br == 0) ? fd1_w : ((br == 1) ? fd2_w : fd3_w);
    const float* fb = (br == 0) ? fd1_b : ((br == 1) ? fd2_b : fd3_b);
    bool v0 = m16 < valid0;
    bool v1 = m16 < 8;
    const float* Arow0 = pooled + (size_t)(row0 + (v0 ? m16 : 0)) * SHOT_CH;
    const float* Arow1 = pooled + (size_t)(32   + (v1 ? m16 : 0)) * SHOT_CH;
    int nCol = nt * 16 + m16;
    const float* Brow = fw + (size_t)nCol * SHOT_CH;
    v8f acc0 = {}, acc1 = {};
    for (int kb = 0; kb < SHOT_CH; kb += 32) {      // 253 steps
        __builtin_prefetch(Brow + kb + 128, 0, 0);
        v16h b = load_b_frag_nt(Brow, kb, hi);
        v4f a0 = *(const v4f*)(Arow0 + kb + hi * 8);
        v4f a1 = *(const v4f*)(Arow0 + kb + hi * 8 + 4);
        v4f a2 = *(const v4f*)(Arow0 + kb + 16 + hi * 8);
        v4f a3 = *(const v4f*)(Arow0 + kb + 16 + hi * 8 + 4);
        if (!v0) { a0 = (v4f){}; a1 = (v4f){}; a2 = (v4f){}; a3 = (v4f){}; }
        v16h af0 = pack16(a0, a1, a2, a3);
        acc0 = __builtin_amdgcn_wmma_f32_16x16x32_f16(false, af0, false, b,
                                                      (short)0, acc0, false, false);
        if (twoTiles) {                  // uniform branch: EXEC still all-ones
            v4f c0 = *(const v4f*)(Arow1 + kb + hi * 8);
            v4f c1 = *(const v4f*)(Arow1 + kb + hi * 8 + 4);
            v4f c2 = *(const v4f*)(Arow1 + kb + 16 + hi * 8);
            v4f c3 = *(const v4f*)(Arow1 + kb + 16 + hi * 8 + 4);
            if (!v1) { c0 = (v4f){}; c1 = (v4f){}; c2 = (v4f){}; c3 = (v4f){}; }
            v16h af1 = pack16(c0, c1, c2, c3);
            acc1 = __builtin_amdgcn_wmma_f32_16x16x32_f16(false, af1, false, b,
                                                          (short)0, acc1, false, false);
        }
    }
    float bias = fb[nCol];
    #pragma unroll
    for (int v = 0; v < 8; ++v) {
        int mLocal = v + hi * 8;
        if (mLocal < valid0) {
            float val = acc0[v] + bias;
            h[(size_t)(row0 + mLocal) * IN_CH + nCol] = val > 0.0f ? val : 0.0f;
        }
        if (twoTiles && mLocal < 8) {
            float val = acc1[v] + bias;
            h[(size_t)(32 + mLocal) * IN_CH + nCol] = val > 0.0f ? val : 0.0f;
        }
    }
}

// ---- Kernel E: per-row LayerNorm over IN_CH -> cross[48,1024] -----------
__global__ void k_layernorm(const float* __restrict__ h,
                            const float* __restrict__ g1, const float* __restrict__ g2,
                            const float* __restrict__ g3,
                            const float* __restrict__ b1, const float* __restrict__ b2,
                            const float* __restrict__ b3,
                            float* __restrict__ cross) {
    __shared__ float s1[256], s2[256];
    int r = blockIdx.x, t = threadIdx.x;
    const float* g = (r < 16) ? g1 : ((r < 40) ? g2 : g3);
    const float* b = (r < 16) ? b1 : ((r < 40) ? b2 : b3);
    const float* hr = h + (size_t)r * IN_CH;
    float x0 = hr[t], x1 = hr[t + 256], x2 = hr[t + 512], x3 = hr[t + 768];
    s1[t] = x0 + x1 + x2 + x3;
    s2[t] = x0*x0 + x1*x1 + x2*x2 + x3*x3;
    __syncthreads();
    for (int off = 128; off > 0; off >>= 1) {
        if (t < off) { s1[t] += s1[t + off]; s2[t] += s2[t + off]; }
        __syncthreads();
    }
    float mu  = s1[0] * (1.0f / IN_CH);
    float var = s2[0] * (1.0f / IN_CH) - mu * mu;
    float ri  = rsqrtf(var + 1e-5f);
    float* cr = cross + (size_t)r * IN_CH;
    cr[t]       = (x0 - mu) * ri * g[t]       + b[t];
    cr[t + 256] = (x1 - mu) * ri * g[t + 256] + b[t + 256];
    cr[t + 512] = (x2 - mu) * ri * g[t + 512] + b[t + 512];
    cr[t + 768] = (x3 - mu) * ri * g[t + 768] + b[t + 768];
}

// ---- Kernel F: out = seq + cross[min(t/170, 47)]  (float4 streaming) ----
// Output is written once and never re-read: non-temporal store.
__global__ void k_merge(const float* __restrict__ seq, const float* __restrict__ cross,
                        float* __restrict__ out) {
    int t = blockIdx.x;
    int m = t / 170; if (m > 47) m = 47;
    const v4f* s = (const v4f*)(seq + (size_t)t * IN_CH);
    const v4f* c = (const v4f*)(cross + (size_t)m * IN_CH);
    v4f* o = (v4f*)(out + (size_t)t * IN_CH);
    v4f val = s[threadIdx.x] + c[threadIdx.x];
    __builtin_nontemporal_store(val, o + threadIdx.x);
}

extern "C" void kernel_launch(void* const* d_in, const int* in_sizes, int n_in,
                              void* d_out, int out_size, void* d_ws, size_t ws_size,
                              hipStream_t stream) {
    const float* seq   = (const float*)d_in[0];
    const float* w_up  = (const float*)d_in[1];
    const float* b_up  = (const float*)d_in[2];
    const float* w11 = (const float*)d_in[3];  const float* b11 = (const float*)d_in[4];
    const float* w12 = (const float*)d_in[5];  const float* b12 = (const float*)d_in[6];
    const float* w13 = (const float*)d_in[7];  const float* b13 = (const float*)d_in[8];
    const float* w14 = (const float*)d_in[9];  const float* b14 = (const float*)d_in[10];
    const float* fd1_w = (const float*)d_in[11]; const float* fd1_b = (const float*)d_in[12];
    const float* ln1_g = (const float*)d_in[13]; const float* ln1_b = (const float*)d_in[14];
    const float* w21 = (const float*)d_in[15]; const float* b21 = (const float*)d_in[16];
    const float* w22 = (const float*)d_in[17]; const float* b22 = (const float*)d_in[18];
    const float* w23 = (const float*)d_in[19]; const float* b23 = (const float*)d_in[20];
    const float* w24 = (const float*)d_in[21]; const float* b24 = (const float*)d_in[22];
    const float* fd2_w = (const float*)d_in[23]; const float* fd2_b = (const float*)d_in[24];
    const float* ln2_g = (const float*)d_in[25]; const float* ln2_b = (const float*)d_in[26];
    const float* w31 = (const float*)d_in[27]; const float* b31 = (const float*)d_in[28];
    const float* w32 = (const float*)d_in[29]; const float* b32 = (const float*)d_in[30];
    const float* w33 = (const float*)d_in[31]; const float* b33 = (const float*)d_in[32];
    const float* w34 = (const float*)d_in[33]; const float* b34 = (const float*)d_in[34];
    const float* fd3_w = (const float*)d_in[35]; const float* fd3_b = (const float*)d_in[36];
    const float* ln3_g = (const float*)d_in[37]; const float* ln3_b = (const float*)d_in[38];

    float* ws = (float*)d_ws;
    float* meanrows = ws;                                   // 48*1024
    float* shot     = meanrows + NBINS * IN_CH;             // 48*8096
    float* pooled   = shot + NBINS * SHOT_CH;               // 48*8096
    float* hbuf     = pooled + NBINS * SHOT_CH;             // 48*1024
    float* cross    = hbuf + NBINS * IN_CH;                 // 48*1024
    float* out      = (float*)d_out;

    k_binmean<<<dim3(NBINS, 4), 256, 0, stream>>>(seq, meanrows);
    k_shot_gemm<<<dim3(SHOT_CH / 16), 32, 0, stream>>>(meanrows, w_up, b_up, shot);
    k_conv_pool<<<dim3(64, 3), 128, 0, stream>>>(shot,
        w11, w12, w13, w14, b11, b12, b13, b14,
        w21, w22, w23, w24, b21, b22, b23, b24,
        w31, w32, w33, w34, b31, b32, b33, b34, pooled);
    k_fd_gemm<<<dim3(IN_CH / 16, 3), 32, 0, stream>>>(pooled,
        fd1_w, fd2_w, fd3_w, fd1_b, fd2_b, fd3_b, hbuf);
    k_layernorm<<<NBINS, 256, 0, stream>>>(hbuf, ln1_g, ln2_g, ln3_g,
                                           ln1_b, ln2_b, ln3_b, cross);
    k_merge<<<SEQ_LEN, 256, 0, stream>>>(seq, cross, out);
}
